// Swin_18348100289215
// MI455X (gfx1250) — matrix operand order repaired and checked by reading, not compiled, not run on hardware
//
#include <hip/hip_runtime.h>
#include <hip/hip_bf16.h>

// ---------------------------------------------------------------------------
// CDNA5 (gfx1250) Swin forward. bf16 WMMA GEMMs (v_wmma_f32_16x16x32_bf16),
// f32 LN/softmax/residuals. Wave32 everywhere.
// ---------------------------------------------------------------------------

typedef __bf16 bf16_t;
typedef __attribute__((ext_vector_type(16))) __bf16 v16bf;
typedef __attribute__((ext_vector_type(8)))  float  v8f;
typedef __attribute__((ext_vector_type(4)))  unsigned int uint4v;

union FragU { v16bf bf; uint4v u[2]; };

__device__ __forceinline__ v8f v8zero() {
  v8f z = {0.f,0.f,0.f,0.f,0.f,0.f,0.f,0.f};
  return z;
}

// A-fragment (16x32 bf16): lane l -> row (l&15); lanes 0-15 hold K = k0+{0..7, 16..23},
// lanes 16-31 hold K = k0+{8..15, 24..31}  (ISA 7.12.2, 16-bit A layout)
__device__ __forceinline__ v16bf load_fragA(const bf16_t* base, int stride, int row0, int k0, int lane) {
  int r = lane & 15;
  int hi = (lane >> 4) & 1;
  const bf16_t* p = base + (size_t)(row0 + r) * stride + k0 + hi * 8;
  FragU f;
  f.u[0] = *(const uint4v*)(p);
  f.u[1] = *(const uint4v*)(p + 16);
  return f.bf;
}

// B-fragment (32x16 bf16): lane l -> col (l&15); lanes 0-15 hold K = k0+{0..7, 16..23},
// lanes 16-31 hold K = k0+{8..15, 24..31} (matches A layout). "stride" is the
// per-column stride (matrix stored col-major along K, i.e. W is (N,K) row-major).
__device__ __forceinline__ v16bf load_fragB(const bf16_t* base, int stride, int col0, int k0, int lane) {
  int c = lane & 15;
  int hi = (lane >> 4) & 1;
  const bf16_t* p = base + (size_t)(col0 + c) * stride + k0 + hi * 8;
  FragU f;
  f.u[0] = *(const uint4v*)(p);
  f.u[1] = *(const uint4v*)(p + 16);
  return f.bf;
}

__device__ __forceinline__ v8f wmma_bf16(v16bf a, v16bf b, v8f c) {
  return __builtin_amdgcn_wmma_f32_16x16x32_bf16(false, a, false, b, (short)0, c, false, false);
}

// ---------------------------------------------------------------------------
// f32 -> bf16 convert
__global__ void k_cvt(const float* __restrict__ src, bf16_t* __restrict__ dst, long n) {
  long i = (long)blockIdx.x * blockDim.x + threadIdx.x;
  if (i < n) dst[i] = (bf16_t)src[i];
}

// ---------------------------------------------------------------------------
// Patch embed: conv4x4 s4 (NCHW in, HWIO weight) + bias + LN -> X (gap layout rows 0..55)
__global__ void k_patch_embed(const float* __restrict__ x, const float* __restrict__ wgt,
                              const float* __restrict__ bias, const float* __restrict__ g,
                              const float* __restrict__ bn, float* __restrict__ X, int Hcap) {
  int pix = blockIdx.x;                       // B*56*56
  int b = pix / (56 * 56);
  int rem = pix - b * (56 * 56);
  int ph = rem / 56, pw = rem - (rem / 56) * 56;
  int co = threadIdx.x;                       // 96 threads
  float v = bias[co];
  for (int cin = 0; cin < 3; cin++)
    for (int i = 0; i < 4; i++)
      for (int j = 0; j < 4; j++)
        v += x[(((size_t)b * 3 + cin) * 224 + (ph * 4 + i)) * 224 + (pw * 4 + j)]
           * wgt[(((size_t)i * 4 + j) * 3 + cin) * 96 + co];
  __shared__ float vals[96];
  vals[co] = v;
  __syncthreads();
  float s = 0.f, q = 0.f;
  for (int j = 0; j < 96; j++) { float u = vals[j]; s += u; q += u * u; }
  float mean = s * (1.f / 96.f);
  float rstd = rsqrtf(q * (1.f / 96.f) - mean * mean + 1e-5f);
  X[(((size_t)b * Hcap + ph) * 56 + pw) * 96 + co] = (v - mean) * rstd * g[co] + bn[co];
}

// ---------------------------------------------------------------------------
// Fill prompt rows: X[b, Hbase+r, w, c] = prompt[r, w, c]
__global__ void k_prompt(const float* __restrict__ pr, float* __restrict__ X,
                         int Bn, int Hbase, int Hcap, int W, int C) {
  long idx = (long)blockIdx.x * blockDim.x + threadIdx.x;
  long tot = (long)Bn * 7 * W * C;
  if (idx >= tot) return;
  int c = (int)(idx % C);
  long t = idx / C;
  int w = (int)(t % W); t /= W;
  int r = (int)(t % 7);
  int b = (int)(t / 7);
  X[(((size_t)b * Hcap + Hbase + r) * W + w) * C + c] = pr[((size_t)r * W + w) * C + c];
}

// ---------------------------------------------------------------------------
// LN1 fused with roll(-shift) + window partition.  One wave per destination token.
__global__ void k_ln_win(const float* __restrict__ X, const float* __restrict__ g,
                         const float* __restrict__ b, bf16_t* __restrict__ out,
                         int Mw, int C, int H, int W, int Hcap, int shift) {
  int t = blockIdx.x * 8 + (threadIdx.x >> 5);
  if (t >= Mw) return;
  int lane = threadIdx.x & 31;
  int win = t / 49, pos = t - win * 49;
  int r = pos / 7, c = pos - r * 7;
  int nWw = W / 7, nWh = H / 7, wpi = nWh * nWw;
  int bb = win / wpi, wi = win - bb * wpi;
  int iH = wi / nWw, iW = wi - iH * nWw;
  int h = iH * 7 + r + shift; if (h >= H) h -= H;
  int w = iW * 7 + c + shift; if (w >= W) w -= W;
  const float* src = X + (((size_t)bb * Hcap + h) * W + w) * (size_t)C;
  float s = 0.f, q = 0.f;
  for (int j = lane; j < C; j += 32) { float v = src[j]; s += v; q += v * v; }
  for (int o = 16; o; o >>= 1) { s += __shfl_xor(s, o, 32); q += __shfl_xor(q, o, 32); }
  float mean = s / C;
  float rstd = rsqrtf(q / C - mean * mean + 1e-5f);
  bf16_t* dst = out + (size_t)t * C;
  for (int j = lane; j < C; j += 32)
    dst[j] = (bf16_t)((src[j] - mean) * rstd * g[j] + b[j]);
}

// ---------------------------------------------------------------------------
// Plain LN over gap-layout X -> compact bf16.  One wave per row.
__global__ void k_ln(const float* __restrict__ X, const float* __restrict__ g,
                     const float* __restrict__ b, bf16_t* __restrict__ out,
                     int M, int C, int HW, long gapC) {
  int row = blockIdx.x * 8 + (threadIdx.x >> 5);
  if (row >= M) return;
  int lane = threadIdx.x & 31;
  const float* src = X + (size_t)row * C + (gapC ? (size_t)(row / HW) * gapC : 0);
  float s = 0.f, q = 0.f;
  for (int j = lane; j < C; j += 32) { float v = src[j]; s += v; q += v * v; }
  for (int o = 16; o; o >>= 1) { s += __shfl_xor(s, o, 32); q += __shfl_xor(q, o, 32); }
  float mean = s / C;
  float rstd = rsqrtf(q / C - mean * mean + 1e-5f);
  bf16_t* dst = out + (size_t)row * C;
  for (int j = lane; j < C; j += 32)
    dst[j] = (bf16_t)((src[j] - mean) * rstd * g[j] + b[j]);
}

// ---------------------------------------------------------------------------
// Patch-merge gather (x0,x1,x2,x3 concat) + LN(4C) -> bf16.  One wave per merged token.
__global__ void k_ln_merge(const float* __restrict__ X, const float* __restrict__ g,
                           const float* __restrict__ b, bf16_t* __restrict__ out,
                           int Mm, int Cin, int Hp, int Wp, int W2) {
  int t = blockIdx.x * 8 + (threadIdx.x >> 5);
  if (t >= Mm) return;
  int lane = threadIdx.x & 31;
  int H2 = Hp / 2;
  int hw2 = H2 * W2;
  int bb = t / hw2;
  int rem = t - bb * hw2;
  int h2 = rem / W2, w2 = rem - h2 * W2;
  size_t base = (size_t)bb * Hp * Wp * Cin;
  int C4 = 4 * Cin;
  float s = 0.f, q = 0.f;
  for (int j = lane; j < C4; j += 32) {
    int sel = j / Cin, cc = j - sel * Cin;
    int dh = sel & 1, dw = sel >> 1;
    float v = X[base + (((size_t)(2 * h2 + dh)) * Wp + (2 * w2 + dw)) * Cin + cc];
    s += v; q += v * v;
  }
  for (int o = 16; o; o >>= 1) { s += __shfl_xor(s, o, 32); q += __shfl_xor(q, o, 32); }
  float mean = s / C4;
  float rstd = rsqrtf(q / C4 - mean * mean + 1e-5f);
  bf16_t* dst = out + (size_t)t * C4;
  for (int j = lane; j < C4; j += 32) {
    int sel = j / Cin, cc = j - sel * Cin;
    int dh = sel & 1, dw = sel >> 1;
    float v = X[base + (((size_t)(2 * h2 + dh)) * Wp + (2 * w2 + dw)) * Cin + cc];
    dst[j] = (bf16_t)((v - mean) * rstd * g[j] + b[j]);
  }
}

// ---------------------------------------------------------------------------
// Generic WMMA GEMM: out(MxN) = A(MxK,bf16) * W(N,K,bf16)^T [+bias][gelu][+res]
// One wave per 16x16 tile, K-loop in steps of 32.
// res/outF use gap-layout offset m*N + (m/HW)*gapC; outB is compact bf16.
__global__ void __launch_bounds__(256)
k_gemm(const bf16_t* __restrict__ A, const bf16_t* __restrict__ Wt,
       const float* __restrict__ bias, const float* __restrict__ res,
       float* __restrict__ outF, bf16_t* __restrict__ outB,
       int M, int N, int K, int HW, long gapC, int gelu) {
  int tilesN = N >> 4;
  long tiles = (long)(M >> 4) * tilesN;
  long tile = (long)blockIdx.x * 8 + (threadIdx.x >> 5);
  if (tile >= tiles) return;
  int tn = (int)(tile % tilesN);
  int tm = (int)(tile / tilesN);
  int lane = threadIdx.x & 31;
  const bf16_t* arow = A + (size_t)tm * 16 * K;
  const bf16_t* brow = Wt + (size_t)tn * 16 * K;
  v8f acc = v8zero();
  for (int k0 = 0; k0 < K; k0 += 32) {
    v16bf af = load_fragA(arow, K, 0, k0, lane);
    v16bf bf = load_fragB(brow, K, 0, k0, lane);
    acc = wmma_bf16(af, bf, acc);
  }
  int n = tn * 16 + (lane & 15);
  int addm = (lane >> 4) * 8;
  float bv = bias ? bias[n] : 0.f;
  for (int i = 0; i < 8; i++) {
    int m = tm * 16 + i + addm;
    float v = acc[i] + bv;
    if (gelu) v = 0.5f * v * (1.f + erff(v * 0.70710678118654752f));
    size_t off = (size_t)m * N + n;
    size_t offx = gapC ? off + (size_t)(m / HW) * gapC : off;
    if (res) v += res[offx];
    if (outB) outB[off] = (bf16_t)v;
    else      outF[offx] = v;
  }
}

// ---------------------------------------------------------------------------
// Windowed attention: one 128-thread WG per (window, head).  N=49 padded to 64.
__global__ void __launch_bounds__(128)
k_attn(const bf16_t* __restrict__ qkv, const float* __restrict__ rpb,
       bf16_t* __restrict__ out, int C, int nH, int H, int W, int shift, int wpi) {
  int win = blockIdx.x;
  int head = blockIdx.y;
  int S3 = 3 * C;
  const bf16_t* base = qkv + (size_t)win * 49 * S3;
  int qo = head * 32, ko = C + head * 32, vo = 2 * C + head * 32;

  __shared__ float  Sm[64 * 64];
  __shared__ bf16_t Qs[64 * 32];
  __shared__ bf16_t Ks[64 * 32];
  __shared__ bf16_t Vt[32 * 64];
  __shared__ bf16_t Pm[64 * 64];

  int tid = threadIdx.x;
  for (int i = tid; i < 64 * 32; i += 128) {
    int j = i >> 5, d = i & 31;
    bf16_t q = (bf16_t)0.f, k = (bf16_t)0.f, v = (bf16_t)0.f;
    if (j < 49) {
      const bf16_t* row = base + (size_t)j * S3;
      q = row[qo + d]; k = row[ko + d]; v = row[vo + d];
    }
    Qs[i] = q; Ks[i] = k; Vt[d * 64 + j] = v;
  }
  __syncthreads();

  int wv = tid >> 5, lane = tid & 31;
  int wi = win % wpi;
  int nWw = W / 7;
  int iH = wi / nWw, iW = wi - iH * nWw;
  const float scale = 0.17677669529663687f;  // 32^-0.5

  // S = Q K^T (16 tiles, wave wv owns row strip 16*wv)
  v16bf aq = load_fragA((const bf16_t*)Qs, 32, wv * 16, 0, lane);
  for (int jt = 0; jt < 4; jt++) {
    v16bf bk = load_fragB((const bf16_t*)Ks, 32, jt * 16, 0, lane);
    v8f acc = v8zero();
    acc = wmma_bf16(aq, bk, acc);
    int nl = lane & 15, addm = (lane >> 4) * 8;
    for (int i = 0; i < 8; i++) {
      int qi = wv * 16 + i + addm;
      int kj = jt * 16 + nl;
      if (qi < 49 && kj < 49) {
        int r1 = qi / 7, c1 = qi - r1 * 7;
        int r2 = kj / 7, c2 = kj - r2 * 7;
        float v = acc[i] * scale + rpb[((r1 - r2 + 6) * 13 + (c1 - c2 + 6)) * nH + head];
        if (shift) {
          int ha = iH * 7 + r1, wa = iW * 7 + c1;
          int hb = iH * 7 + r2, wb = iW * 7 + c2;
          int ra = (ha < H - 7) ? 0 : ((ha < H - shift) ? 1 : 2);
          int ca = (wa < W - 7) ? 0 : ((wa < W - shift) ? 1 : 2);
          int rb = (hb < H - 7) ? 0 : ((hb < H - shift) ? 1 : 2);
          int cb = (wb < W - 7) ? 0 : ((wb < W - shift) ? 1 : 2);
          if (ra * 3 + ca != rb * 3 + cb) v -= 100.f;
        }
        Sm[qi * 64 + kj] = v;
      }
    }
  }
  __syncthreads();

  // Softmax rows (f32), emit bf16 P with zero padding
  for (int row = tid; row < 64; row += 128) {
    if (row < 49) {
      float mx = -1e30f;
      for (int c2 = 0; c2 < 49; c2++) mx = fmaxf(mx, Sm[row * 64 + c2]);
      float sum = 0.f;
      for (int c2 = 0; c2 < 49; c2++) {
        float e = __expf(Sm[row * 64 + c2] - mx);
        Sm[row * 64 + c2] = e; sum += e;
      }
      float inv = 1.f / sum;
      for (int c2 = 0; c2 < 64; c2++)
        Pm[row * 64 + c2] = (c2 < 49) ? (bf16_t)(Sm[row * 64 + c2] * inv) : (bf16_t)0.f;
    } else {
      for (int c2 = 0; c2 < 64; c2++) Pm[row * 64 + c2] = (bf16_t)0.f;
    }
  }
  __syncthreads();

  // out = P V  (wave wv: rows 16*wv, two 16-wide d tiles, K=64 in two steps)
  for (int dt = 0; dt < 2; dt++) {
    v8f acc = v8zero();
    for (int k0 = 0; k0 < 64; k0 += 32) {
      v16bf pa = load_fragA((const bf16_t*)Pm, 64, wv * 16, k0, lane);
      v16bf vb = load_fragB((const bf16_t*)Vt, 64, dt * 16, k0, lane);
      acc = wmma_bf16(pa, vb, acc);
    }
    int nl = lane & 15, addm = (lane >> 4) * 8;
    for (int i = 0; i < 8; i++) {
      int qi = wv * 16 + i + addm;
      if (qi < 49)
        out[((size_t)win * 49 + qi) * C + head * 32 + dt * 16 + nl] = (bf16_t)acc[i];
    }
  }
}

// ---------------------------------------------------------------------------
// Window reverse + roll(+shift) + residual add into gap-layout X.
__global__ void k_winrev_add(float* __restrict__ X, const float* __restrict__ P,
                             int M, int C, int H, int W, int Hcap, int shift) {
  long idx = (long)blockIdx.x * blockDim.x + threadIdx.x;
  if (idx >= (long)M * C) return;
  int c = (int)(idx % C);
  int m = (int)(idx / C);
  int HWl = H * W;
  int bb = m / HWl;
  int rem = m - bb * HWl;
  int h = rem / W, w = rem - (rem / W) * W;
  int h2 = h - shift; if (h2 < 0) h2 += H;
  int w2 = w - shift; if (w2 < 0) w2 += W;
  int nWw = W / 7, nWh = H / 7;
  long t = (((long)bb * nWh + h2 / 7) * nWw + w2 / 7) * 49 + (h2 % 7) * 7 + (w2 % 7);
  X[(((size_t)bb * Hcap + h) * W + w) * C + c] += P[t * C + c];
}

// ---------------------------------------------------------------------------
// Strided copy (gap layout -> compact feats in d_out)
__global__ void k_copy_feat(const float* __restrict__ X, float* __restrict__ out,
                            int H, int W, int C, int Hcap) {
  long idx = (long)blockIdx.x * blockDim.x + threadIdx.x;
  long tot = (long)16 * H * W * C;
  if (idx >= tot) return;
  int c = (int)(idx % C);
  long t = idx / C;
  int w = (int)(t % W); t /= W;
  int h = (int)(t % H);
  int b = (int)(t / H);
  out[idx] = X[(((size_t)b * Hcap + h) * W + w) * C + c];
}

// ===========================================================================
extern "C" void kernel_launch(void* const* d_in, const int* in_sizes, int n_in,
                              void* d_out, int out_size, void* d_ws, size_t ws_size,
                              hipStream_t stream) {
  (void)in_sizes; (void)n_in; (void)out_size; (void)ws_size;
  static const int DEPTH[4] = {2, 2, 6, 2};
  static const int HH[4]   = {63, 35, 21, 14};
  static const int WWD[4]  = {56, 28, 14, 7};
  static const int CCD[4]  = {96, 192, 384, 768};
  static const int NHEAD[4]= {3, 6, 12, 24};
  static const int HCAP[4] = {70, 42, 28, 14};
  const int Bn = 16;

  // ---- unpack inputs (recursive insertion order of setup_inputs dict) ----
  int idx = 0;
  auto nxt = [&]() { return (const float*)d_in[idx++]; };
  const float* xin = nxt();
  const float* pew = nxt();
  const float* peb = nxt();
  const float* peg = nxt();
  const float* pen = nxt();
  const float* prom[4];
  for (int i = 0; i < 4; i++) prom[i] = nxt();
  const float* mng[4] = {0, 0, 0, 0};
  const float* mnb[4] = {0, 0, 0, 0};
  const float* mwf[4] = {0, 0, 0, 0};
  const float* bw[4][6][13];
  for (int i = 0; i < 4; i++) {
    if (i > 0) { mng[i] = nxt(); mnb[i] = nxt(); mwf[i] = nxt(); }
    for (int d = 0; d < DEPTH[i]; d++)
      for (int p = 0; p < 13; p++) bw[i][d][p] = nxt();
  }
  // block param order: 0 n1g, 1 n1b, 2 qkv_w, 3 qkv_b, 4 proj_w, 5 proj_b,
  //                    6 rpb, 7 n2g, 8 n2b, 9 fc1_w, 10 fc1_b, 11 fc2_w, 12 fc2_b

  // ---- workspace carve ----
  char* wsp = (char*)d_ws;
  auto alloc = [&](size_t bytes) -> void* {
    void* p = (void*)wsp;
    wsp += (bytes + 255) & ~(size_t)255;
    return p;
  };
  float*  X    = (float*)alloc((size_t)6021120 * 4);          // gap-layout features
  bf16_t* XM   = (bf16_t*)alloc((size_t)6021120 * 2);         // merged LN out
  bf16_t* ABF  = (bf16_t*)alloc((size_t)5419008 * 2);         // LN1/LN2 bf16 tokens
  bf16_t* QKV  = (bf16_t*)alloc((size_t)16257024 * 2);        // qkv bf16
  bf16_t* ATT  = (bf16_t*)alloc((size_t)5419008 * 2);         // attention out bf16
  float*  PROJ = (float*)alloc((size_t)5419008 * 4);          // proj out f32 (window order)
  bf16_t* H1   = (bf16_t*)alloc((size_t)21676032 * 2);        // fc1 out bf16

  auto cvt = [&](const float* src, size_t n) -> const bf16_t* {
    bf16_t* dst = (bf16_t*)alloc(n * sizeof(bf16_t));
    k_cvt<<<(unsigned)((n + 255) / 256), 256, 0, stream>>>(src, dst, (long)n);
    return dst;
  };

  const bf16_t* mwb[4] = {0, 0, 0, 0};
  const bf16_t* qkvw[4][6]; const bf16_t* projw[4][6];
  const bf16_t* fc1w[4][6]; const bf16_t* fc2w[4][6];
  for (int i = 0; i < 4; i++) {
    int C = CCD[i];
    if (i > 0) mwb[i] = cvt(mwf[i], (size_t)C * 4 * CCD[i - 1]);
    for (int d = 0; d < DEPTH[i]; d++) {
      qkvw[i][d] = cvt(bw[i][d][2],  (size_t)3 * C * C);
      projw[i][d]= cvt(bw[i][d][4],  (size_t)C * C);
      fc1w[i][d] = cvt(bw[i][d][9],  (size_t)4 * C * C);
      fc2w[i][d] = cvt(bw[i][d][11], (size_t)4 * C * C);
    }
  }

  auto gemm = [&](const bf16_t* A, const bf16_t* Wt, const float* bias, const float* res,
                  float* oF, bf16_t* oB, int M, int N, int K, int HW, long gap, int gelu) {
    long tiles = (long)(M / 16) * (N / 16);
    k_gemm<<<(unsigned)((tiles + 7) / 8), 256, 0, stream>>>(A, Wt, bias, res, oF, oB,
                                                            M, N, K, HW, gap, gelu);
  };

  float* outp = (float*)d_out;
  long outOff = 0;

  // ---- patch embed + stage 0 prompt ----
  k_patch_embed<<<Bn * 56 * 56, 96, 0, stream>>>(xin, pew, peb, peg, pen, X, HCAP[0]);
  k_prompt<<<(unsigned)(((long)Bn * 7 * 56 * 96 + 255) / 256), 256, 0, stream>>>(
      prom[0], X, Bn, 56, HCAP[0], 56, 96);

  for (int i = 0; i < 4; i++) {
    int H = HH[i], W = WWD[i], C = CCD[i], nh = NHEAD[i], Hcap = HCAP[i];
    int M = Bn * H * W;
    int HWl = H * W;
    long gap = (long)(Hcap - H) * W * C;
    int nWh = H / 7, nWw = W / 7;
    int totalWin = Bn * nWh * nWw;

    if (i > 0) {
      int Hb = HH[i - 1], Hcp = HCAP[i - 1], Wp = WWD[i - 1], Cp = CCD[i - 1];
      k_prompt<<<(unsigned)(((long)Bn * 7 * Wp * Cp + 255) / 256), 256, 0, stream>>>(
          prom[i], X, Bn, Hb, Hcp, Wp, Cp);
      int Mm = Bn * (Hcp / 2) * (Wp / 2);
      k_ln_merge<<<(unsigned)((Mm + 7) / 8), 256, 0, stream>>>(
          X, mng[i], mnb[i], XM, Mm, Cp, Hcp, Wp, Wp / 2);
      gemm(XM, mwb[i], nullptr, nullptr, X, nullptr, Mm, C, 4 * Cp, HWl, gap, 0);
    }

    for (int d = 0; d < DEPTH[i]; d++) {
      int shift = (d & 1) ? 3 : 0;
      k_ln_win<<<(unsigned)((M + 7) / 8), 256, 0, stream>>>(
          X, bw[i][d][0], bw[i][d][1], ABF, M, C, H, W, Hcap, shift);
      gemm(ABF, qkvw[i][d], bw[i][d][3], nullptr, nullptr, QKV, M, 3 * C, C, 1, 0, 0);
      k_attn<<<dim3(totalWin, nh), 128, 0, stream>>>(
          QKV, bw[i][d][6], ATT, C, nh, H, W, shift, nWh * nWw);
      gemm(ATT, projw[i][d], bw[i][d][5], nullptr, PROJ, nullptr, M, C, C, 1, 0, 0);
      {
        long tot = (long)M * C;
        k_winrev_add<<<(unsigned)((tot + 255) / 256), 256, 0, stream>>>(
            X, PROJ, M, C, H, W, Hcap, shift);
      }
      k_ln<<<(unsigned)((M + 7) / 8), 256, 0, stream>>>(
          X, bw[i][d][7], bw[i][d][8], ABF, M, C, HWl, gap);
      gemm(ABF, fc1w[i][d], bw[i][d][10], nullptr, nullptr, H1, M, 4 * C, C, 1, 0, 1);
      gemm(H1, fc2w[i][d], bw[i][d][12], X, X, nullptr, M, C, 4 * C, HWl, gap, 0);
    }

    long tot = (long)M * C;
    k_copy_feat<<<(unsigned)((tot + 255) / 256), 256, 0, stream>>>(
        X, outp + outOff, H, W, C, Hcap);
    outOff += tot;
  }
}
